// BERTWeightedLayer_49211735278160
// MI455X (gfx1250) — matrix operand order repaired and checked by reading, not compiled, not run on hardware
//
#include <hip/hip_runtime.h>
#include <math.h>

typedef __attribute__((ext_vector_type(16))) __bf16 v16bf;
typedef __attribute__((ext_vector_type(8)))  float  v8f;
typedef __attribute__((ext_vector_type(4)))  unsigned int u32x4;
typedef int gv4i __attribute__((vector_size(16)));   // matches builtin's V4i param type
typedef __bf16 bf16;

#define HIDN 768
#define SEQ  512
#define BATCH 4
#define NH   12
#define HD   64
#define INTER 3072
#define ROWS (BATCH*SEQ)   // 2048

// ---------------- async global->LDS staging (CDNA5 ASYNCcnt path) ----------------
__device__ __forceinline__ void async_copy_b128(const bf16* gsrc, bf16* ldst) {
#if __has_builtin(__builtin_amdgcn_global_load_async_to_lds_b128)
  __builtin_amdgcn_global_load_async_to_lds_b128(
      (__attribute__((address_space(1))) gv4i*)(bf16*)gsrc,
      (__attribute__((address_space(3))) gv4i*)ldst, 0, 0);
#else
  *(u32x4*)ldst = *(const u32x4*)gsrc;
#endif
}
__device__ __forceinline__ void wait_async0() {
#if __has_builtin(__builtin_amdgcn_s_wait_asynccnt)
  __builtin_amdgcn_s_wait_asynccnt(0);
#else
  asm volatile("s_wait_asynccnt 0" ::: "memory");
#endif
}

// ---------------- WMMA fragment loaders (bf16, 16x16x32) ----------------
// A fragment (16 rows x 32 K), row-major source with leading dim `ld`:
//   lane L: row = L&15; lanes 0-15 hold K 0..15, lanes 16-31 hold K 16..31 (contiguous)
__device__ __forceinline__ v16bf load_a_frag(const bf16* tile, int ld, int kk, int lane) {
  const int r  = lane & 15;
  const int hs = (lane >> 4) << 4;
  const bf16* p = tile + (size_t)r * ld + kk + hs;
  v16bf f;
#pragma unroll
  for (int j = 0; j < 16; ++j) f[j] = p[j];
  return f;
}
// B fragment (32 K x 16 cols), loaded from Bt = B^T (N-major, ld = K stride):
//   lane L: col = L&15; lanes 0-15 hold K 0..15, lanes 16-31 hold K 16..31 (contiguous)
__device__ __forceinline__ v16bf load_b_frag(const bf16* tileT, int ld, int kk, int lane) {
  const int n  = lane & 15;
  const int hs = (lane >> 4) << 4;
  const bf16* p = tileT + (size_t)n * ld + kk + hs;
  v16bf f;
#pragma unroll
  for (int j = 0; j < 16; ++j) f[j] = p[j];
  return f;
}

#define WMMA_BF16(acc, a, b) \
  acc = __builtin_amdgcn_wmma_f32_16x16x32_bf16(false, (a), false, (b), (short)0, (acc), false, false)

__device__ __forceinline__ float gelu_f(float x) {
  return 0.5f * x * (1.0f + erff(x * 0.70710678118654752f));
}

// ---------------- conversion / transpose / zero ----------------
__global__ void k_cvt(const float* __restrict__ src, bf16* __restrict__ dst, int n) {
  int i = blockIdx.x * blockDim.x + threadIdx.x;
  if (i < n) dst[i] = (bf16)src[i];
}
// dst[c*R + r] = src[r*C + c]  (store W^T so B-fragments are contiguous)
__global__ void k_tr(const float* __restrict__ src, bf16* __restrict__ dst, int R, int C) {
  int i = blockIdx.x * blockDim.x + threadIdx.x;
  if (i < R * C) {
    int r = i / C, c = i - r * C;
    dst[(size_t)c * R + r] = (bf16)src[i];
  }
}
// Wo [H][D][HID] -> Wot [H][HID][D]
__global__ void k_tr_wo(const float* __restrict__ src, bf16* __restrict__ dst) {
  int i = blockIdx.x * blockDim.x + threadIdx.x;
  if (i < NH * HD * HIDN) {
    int h = i / (HD * HIDN);
    int rem = i - h * HD * HIDN;
    int d = rem / HIDN;
    int m = rem - d * HIDN;
    dst[((size_t)h * HIDN + m) * HD + d] = (bf16)src[i];
  }
}
__global__ void k_zero(float* __restrict__ p, int n) {
  int i = blockIdx.x * blockDim.x + threadIdx.x;
  if (i < n) p[i] = 0.0f;
}

// ---------------- QKV projection: [2048x768] @ [768x768] + bias -> bf16 ----------------
// A tile staged once per WG into LDS via async-to-LDS; each wave does 3 N-tiles
// register-blocked on one shared A fragment.
#define ALD 776
__global__ void k_qkv(const bf16* __restrict__ X, const bf16* __restrict__ Wt,
                      const float* __restrict__ bias, bf16* __restrict__ dst) {
  __shared__ bf16 As[16 * ALD];
  const int tid  = threadIdx.x;
  const int lane = tid & 31;
  const int wave = tid >> 5;
  const int mbase = blockIdx.x * 16;

  const bf16* Ag = X + (size_t)mbase * HIDN;
#pragma unroll
  for (int s = 0; s < 6; ++s) {           // 16 rows * 96 chunks(16B) = 1536 chunks
    int i = s * 256 + tid;
    int r = i / 96, c = i - r * 96;
    async_copy_b128(Ag + (size_t)r * HIDN + c * 8, &As[r * ALD + c * 8]);
  }
  wait_async0();
  __syncthreads();

  const int n0 = blockIdx.y * 384 + wave * 48;    // 3 tiles of 16 per wave
  v8f acc[3] = {};
  for (int kk = 0; kk < HIDN; kk += 32) {
    v16bf a = load_a_frag(As, ALD, kk, lane);
#pragma unroll
    for (int t = 0; t < 3; ++t) {
      v16bf b = load_b_frag(Wt + (size_t)(n0 + t * 16) * HIDN, HIDN, kk, lane);
      WMMA_BF16(acc[t], a, b);
    }
  }
  const int col15 = lane & 15;
  const int rbase = mbase + ((lane >> 4) << 3);
#pragma unroll
  for (int t = 0; t < 3; ++t) {
    const int col = n0 + t * 16 + col15;
    const float bv = bias[col];
#pragma unroll
    for (int v = 0; v < 8; ++v)
      dst[(size_t)(rbase + v) * HIDN + col] = (bf16)(acc[t][v] + bv);
  }
}

// ---------------- attention: scores (WMMA) + softmax + P.V ----------------
__global__ void k_attn(const bf16* __restrict__ Q, const bf16* __restrict__ K,
                       const bf16* __restrict__ V, const float* __restrict__ mask,
                       bf16* __restrict__ CTX) {
  __shared__ float sc[16][SEQ];
  __shared__ bf16  pb[16][SEQ];
  __shared__ float red[16][16];
  __shared__ float rowstat[16];
  const int lane  = threadIdx.x & 31;
  const int wave  = threadIdx.x >> 5;
  const int mbase = blockIdx.x * 16;
  const int h = blockIdx.y, b = blockIdx.z;
  const bf16* Qt = Q + ((size_t)(b * SEQ + mbase)) * HIDN + h * HD;

  // scores: each wave 4 N-tiles, K=64 (2 wmma-ksteps), A shared across tiles
  {
    v8f acc[4] = {};
#pragma unroll
    for (int kk = 0; kk < HD; kk += 32) {
      v16bf a = load_a_frag(Qt, HIDN, kk, lane);
#pragma unroll
      for (int t = 0; t < 4; ++t) {
        const int n0 = wave * 64 + t * 16;
        v16bf bb = load_b_frag(K + ((size_t)(b * SEQ + n0)) * HIDN + h * HD, HIDN, kk, lane);
        WMMA_BF16(acc[t], a, bb);
      }
    }
    const int rb = (lane >> 4) << 3;
#pragma unroll
    for (int t = 0; t < 4; ++t) {
      const int col = wave * 64 + t * 16 + (lane & 15);
      const float mval = mask[b * SEQ + col];
#pragma unroll
      for (int v = 0; v < 8; ++v) sc[rb + v][col] = acc[t][v] * 0.125f + mval;
    }
  }
  __syncthreads();

  const int row = threadIdx.x >> 4;
  const int sub = threadIdx.x & 15;
  float pmax = -3.4e38f;
  for (int c = sub; c < SEQ; c += 16) pmax = fmaxf(pmax, sc[row][c]);
  red[row][sub] = pmax;
  __syncthreads();
  if (sub == 0) {
    float m = red[row][0];
    for (int j = 1; j < 16; ++j) m = fmaxf(m, red[row][j]);
    rowstat[row] = m;
  }
  __syncthreads();
  const float m = rowstat[row];
  float psum = 0.0f;
  for (int c = sub; c < SEQ; c += 16) {
    float e = expf(sc[row][c] - m);
    sc[row][c] = e;
    psum += e;
  }
  red[row][sub] = psum;
  __syncthreads();
  if (sub == 0) {
    float s = 0.0f;
    for (int j = 0; j < 16; ++j) s += red[row][j];
    rowstat[row] = 1.0f / s;
  }
  __syncthreads();
  const float inv = rowstat[row];
  for (int c = sub; c < SEQ; c += 16) pb[row][c] = (bf16)(sc[row][c] * inv);
  __syncthreads();

  // ctx[row][d] = sum_s P[row][s] * V[s][d], d = sub + 16j
  float acc4[4] = {0.f, 0.f, 0.f, 0.f};
  const bf16* Vb = V + (size_t)b * SEQ * HIDN + h * HD;
  for (int s = 0; s < SEQ; ++s) {
    float p = (float)pb[row][s];
#pragma unroll
    for (int j = 0; j < 4; ++j)
      acc4[j] += p * (float)Vb[(size_t)s * HIDN + sub + 16 * j];
  }
#pragma unroll
  for (int j = 0; j < 4; ++j)
    CTX[((size_t)(b * SEQ + mbase + row)) * HIDN + h * HD + sub + 16 * j] = (bf16)acc4[j];
}

// ---------------- fused per-head Wo-proj + FFN, intermediate in LDS ----------------
#define XLD 776     // 768 + 8 pad (bank-conflict avoidance)
#define ILD 3080    // 3072 + 8 pad
__global__ void k_ffn(const bf16* __restrict__ CTX, const bf16* __restrict__ Wot,
                      const float* __restrict__ bo, const float* __restrict__ wkp,
                      const bf16* __restrict__ Wit, const float* __restrict__ bi,
                      const bf16* __restrict__ Woutt, const float* __restrict__ wa,
                      float* __restrict__ OUT) {
  __shared__ bf16 xs[16 * XLD];   // x tile  (16 x 768)
  __shared__ bf16 is[16 * ILD];   // inter   (16 x 3072)
  const int lane  = threadIdx.x & 31;
  const int wave  = threadIdx.x >> 5;
  const int mbase = blockIdx.x * 16;
  const int h     = blockIdx.y;
  const int col15 = lane & 15;
  const int rb    = (lane >> 4) << 3;
  const float kp  = wkp[h];

  // phase 0: x = kp * (ctx_h @ Wo_h + bo_h)   (K=64, N=768; 6 tiles/wave, NB=6)
  {
    const bf16* At0 = CTX + (size_t)mbase * HIDN + h * HD;
    const int nb0 = wave * 96;
    v8f acc[6] = {};
#pragma unroll
    for (int kk = 0; kk < HD; kk += 32) {
      v16bf a = load_a_frag(At0, HIDN, kk, lane);
#pragma unroll
      for (int t = 0; t < 6; ++t) {
        v16bf b = load_b_frag(Wot + ((size_t)h * HIDN + nb0 + t * 16) * HD, HD, kk, lane);
        WMMA_BF16(acc[t], a, b);
      }
    }
#pragma unroll
    for (int t = 0; t < 6; ++t) {
      const int c = nb0 + t * 16 + col15;
      const float bv = bo[h * HIDN + c];
#pragma unroll
      for (int v = 0; v < 8; ++v)
        xs[(rb + v) * XLD + c] = (bf16)(kp * (acc[t][v] + bv));
    }
  }
  __syncthreads();

  // phase 1: inter = gelu(x @ Wi + bi)   (K=768, N=3072; 24 tiles/wave in 6 groups of NB=4)
  for (int g = 0; g < 6; ++g) {
    const int base = (wave * 24 + g * 4) * 16;
    v8f acc[4] = {};
    for (int kk = 0; kk < HIDN; kk += 32) {
      v16bf a = load_a_frag(xs, XLD, kk, lane);
#pragma unroll
      for (int t = 0; t < 4; ++t) {
        v16bf b = load_b_frag(Wit + (size_t)(base + t * 16) * HIDN, HIDN, kk, lane);
        WMMA_BF16(acc[t], a, b);
      }
    }
#pragma unroll
    for (int t = 0; t < 4; ++t) {
      const int c = base + t * 16 + col15;
      const float bv = bi[c];
#pragma unroll
      for (int v = 0; v < 8; ++v) {
        float x = acc[t][v] + bv;
        is[(rb + v) * ILD + c] = (bf16)gelu_f(x);
      }
    }
  }
  __syncthreads();

  // phase 2: out += wa * (inter @ Wout)   (K=3072, N=768; 6 tiles/wave, NB=6)
  {
    const float aw = wa[h];
    const int nb0 = wave * 96;
    v8f acc[6] = {};
    for (int kk = 0; kk < INTER; kk += 32) {
      v16bf a = load_a_frag(is, ILD, kk, lane);
#pragma unroll
      for (int t = 0; t < 6; ++t) {
        v16bf b = load_b_frag(Woutt + (size_t)(nb0 + t * 16) * INTER, INTER, kk, lane);
        WMMA_BF16(acc[t], a, b);
      }
    }
#pragma unroll
    for (int t = 0; t < 6; ++t) {
      const int c = nb0 + t * 16 + col15;
#pragma unroll
      for (int v = 0; v < 8; ++v)
        atomicAdd(&OUT[(size_t)(mbase + rb + v) * HIDN + c], aw * acc[t][v]);
    }
  }
}

// ---------------- residual + sum(w_a)*bout + layernorm ----------------
__global__ void k_ln(const float* __restrict__ HS, const float* __restrict__ OUT,
                     const float* __restrict__ bout, const float* __restrict__ wa,
                     const float* __restrict__ gamma, const float* __restrict__ beta,
                     float* __restrict__ Y) {
  __shared__ float r1[256], r2[256];
  const int row = blockIdx.x;
  const int t = threadIdx.x;
  float sw = 0.0f;
  for (int hh = 0; hh < NH; ++hh) sw += wa[hh];
  float vals[3];
  float s = 0.0f, s2 = 0.0f;
#pragma unroll
  for (int j = 0; j < 3; ++j) {
    int c = t + j * 256;
    float v = HS[(size_t)row * HIDN + c] + OUT[(size_t)row * HIDN + c] + sw * bout[c];
    vals[j] = v; s += v; s2 += v * v;
  }
  r1[t] = s; r2[t] = s2;
  __syncthreads();
  for (int off = 128; off > 0; off >>= 1) {
    if (t < off) { r1[t] += r1[t + off]; r2[t] += r2[t + off]; }
    __syncthreads();
  }
  const float mean = r1[0] * (1.0f / HIDN);
  const float var  = r2[0] * (1.0f / HIDN) - mean * mean;
  const float rinv = rsqrtf(var + 1e-12f);
#pragma unroll
  for (int j = 0; j < 3; ++j) {
    int c = t + j * 256;
    Y[(size_t)row * HIDN + c] = gamma[c] * (vals[j] - mean) * rinv + beta[c];
  }
}

// ---------------- host-side orchestration ----------------
extern "C" void kernel_launch(void* const* d_in, const int* in_sizes, int n_in,
                              void* d_out, int out_size, void* d_ws, size_t ws_size,
                              hipStream_t stream) {
  const float* hs   = (const float*)d_in[0];
  const float* mask = (const float*)d_in[1];
  const float* Wq   = (const float*)d_in[2];  const float* bq   = (const float*)d_in[3];
  const float* Wk   = (const float*)d_in[4];  const float* bk   = (const float*)d_in[5];
  const float* Wv   = (const float*)d_in[6];  const float* bv   = (const float*)d_in[7];
  const float* Wo   = (const float*)d_in[8];  const float* bo   = (const float*)d_in[9];
  const float* wkp  = (const float*)d_in[10]; const float* wa   = (const float*)d_in[11];
  const float* Wi   = (const float*)d_in[12]; const float* bi   = (const float*)d_in[13];
  const float* Wout = (const float*)d_in[14]; const float* bout = (const float*)d_in[15];
  const float* gamma = (const float*)d_in[16]; const float* beta = (const float*)d_in[17];
  float* Y = (float*)d_out;
  (void)in_sizes; (void)n_in; (void)out_size; (void)ws_size;

  char* wsb = (char*)d_ws;
  size_t off = 0;
  auto alloc = [&](size_t bytes) -> void* {
    void* p = wsb + off;
    off += (bytes + 255) & ~(size_t)255;
    return p;
  };
  bf16* hsb   = (bf16*)alloc((size_t)ROWS * HIDN * 2);
  bf16* Wqt   = (bf16*)alloc((size_t)HIDN * HIDN * 2);
  bf16* Wkt   = (bf16*)alloc((size_t)HIDN * HIDN * 2);
  bf16* Wvt   = (bf16*)alloc((size_t)HIDN * HIDN * 2);
  bf16* qb    = (bf16*)alloc((size_t)ROWS * HIDN * 2);
  bf16* kb    = (bf16*)alloc((size_t)ROWS * HIDN * 2);
  bf16* vb    = (bf16*)alloc((size_t)ROWS * HIDN * 2);
  bf16* Wot   = (bf16*)alloc((size_t)NH * HIDN * HD * 2);
  bf16* ctxb  = (bf16*)alloc((size_t)ROWS * HIDN * 2);
  bf16* Wit   = (bf16*)alloc((size_t)INTER * HIDN * 2);
  bf16* Woutt = (bf16*)alloc((size_t)HIDN * INTER * 2);
  float* outacc = (float*)alloc((size_t)ROWS * HIDN * 4);

  const int T = 256;
  k_cvt<<<(ROWS * HIDN + T - 1) / T, T, 0, stream>>>(hs, hsb, ROWS * HIDN);
  k_tr<<<(HIDN * HIDN + T - 1) / T, T, 0, stream>>>(Wq, Wqt, HIDN, HIDN);
  k_tr<<<(HIDN * HIDN + T - 1) / T, T, 0, stream>>>(Wk, Wkt, HIDN, HIDN);
  k_tr<<<(HIDN * HIDN + T - 1) / T, T, 0, stream>>>(Wv, Wvt, HIDN, HIDN);
  k_tr_wo<<<(NH * HD * HIDN + T - 1) / T, T, 0, stream>>>(Wo, Wot);
  k_tr<<<(HIDN * INTER + T - 1) / T, T, 0, stream>>>(Wi, Wit, HIDN, INTER);
  k_tr<<<(INTER * HIDN + T - 1) / T, T, 0, stream>>>(Wout, Woutt, INTER, HIDN);
  k_zero<<<(ROWS * HIDN + T - 1) / T, T, 0, stream>>>(outacc, ROWS * HIDN);

  k_qkv<<<dim3(ROWS / 16, 2, 1), T, 0, stream>>>(hsb, Wqt, bq, qb);
  k_qkv<<<dim3(ROWS / 16, 2, 1), T, 0, stream>>>(hsb, Wkt, bk, kb);
  k_qkv<<<dim3(ROWS / 16, 2, 1), T, 0, stream>>>(hsb, Wvt, bv, vb);

  k_attn<<<dim3(SEQ / 16, NH, BATCH), T, 0, stream>>>(qb, kb, vb, mask, ctxb);

  k_ffn<<<dim3(ROWS / 16, NH, 1), T, 0, stream>>>(ctxb, Wot, bo, wkp, Wit, bi, Woutt, wa, outacc);

  k_ln<<<ROWS, T, 0, stream>>>(hs, outacc, bout, wa, gamma, beta, Y);
}
